// LCALSTM_46540265619652
// MI455X (gfx1250) — compile-verified, matrix operand order checked by reading, not run on hardware
//
#include <hip/hip_runtime.h>
#include <math.h>

#define H      512
#define DIN    512
#define NACT   16
#define DICT   8192
#define BATCH  4096
#define NCYC   10
#define DTT    0.6f
#define EPSV   1e-8f
#define NHT    2051
#define NHT_PAD 2112
#define KCAT   1024

#define BM_T   128        // block tile M
#define BN_T   64         // block tile N
#define ASZ    (BM_T * 32)
#define BSZ    (BN_T * 32)

typedef __attribute__((ext_vector_type(16))) __bf16 v16bf;
typedef __attribute__((ext_vector_type(8)))  float  v8f;

union Frag { v16bf v; uint4 q[2]; };

__device__ __forceinline__ unsigned short f2bf(float f) {
  unsigned int u = __float_as_uint(f);
  u += 0x7FFFu + ((u >> 16) & 1u);   // round-to-nearest-even
  return (unsigned short)(u >> 16);
}
__device__ __forceinline__ float sigf(float x) { return 1.0f / (1.0f + __expf(-x)); }

// ---------------------------------------------------------------------------
// CDNA5 async copy: global -> LDS, 16B per lane, tracked by ASYNCcnt.
// ---------------------------------------------------------------------------
__device__ __forceinline__ void async_ld16(unsigned ldsOff,
                                           const unsigned short* gptr) {
  asm volatile("global_load_async_to_lds_b128 %0, %1, off"
               :
               : "v"(ldsOff), "v"((unsigned long long)(uintptr_t)gptr)
               : "memory");
}
__device__ __forceinline__ unsigned lds_off(const unsigned short* p) {
  return (unsigned)(uintptr_t)p;     // low 32 bits of flat LDS addr = LDS offset
}

// Issue async loads for one 128x32 A tile + 64x32 B tile (3 b128 per thread).
__device__ __forceinline__ void issue_tile_async(
    const unsigned short* __restrict__ A, int lda, int bm, int k0,
    const unsigned short* __restrict__ B, int ldb, int bn,
    unsigned short* cA, unsigned short* cB)
{
  const int t = threadIdx.x;
#pragma unroll
  for (int c = 0; c < 2; ++c) {          // A: 128*32 elems = 512 x b128
    int ci  = t + c * 256;
    int row = ci >> 2, col = (ci & 3) * 8;
    async_ld16(lds_off(&cA[row * 32 + col]),
               &A[(size_t)(bm + row) * lda + k0 + col]);
  }
  {                                      // B: 64*32 elems = 256 x b128
    int row = t >> 2, col = (t & 3) * 8;
    async_ld16(lds_off(&cB[row * 32 + col]),
               &B[(size_t)(bn + row) * ldb + k0 + col]);
  }
}

// ---------------------------------------------------------------------------
// Core: 128x64 block tile, K-step 32, double-buffered LDS fed by async loads.
// 256 threads = 8 waves in 4(M) x 2(N) grid; each wave owns 32x32 (4 WMMA).
// A: M x K row-major bf16(u16), B: N x K row-major (computes A @ B^T).
// ---------------------------------------------------------------------------
__device__ __forceinline__ void gemm_core_128x64(
    const unsigned short* __restrict__ A, int lda,
    const unsigned short* __restrict__ B, int ldb,
    int K, int bm, int bn,
    unsigned short* lsA, unsigned short* lsB,   // [2][ASZ], [2][BSZ]
    v8f acc[2][2])
{
  const int t    = threadIdx.x;
  const int lane = t & 31, wv = t >> 5;
  const int mw   = wv & 3,  nw = wv >> 2;
  const int lr   = lane & 15, hi = lane >> 4;
  const int kb   = hi * 8;                 // K-half select, 16-bit A layout

  issue_tile_async(A, lda, bm, 0, B, ldb, bn, lsA, lsB);

  int buf = 0;
  for (int k0 = 0; k0 < K; k0 += 32, buf ^= 1) {
    unsigned short* cA = lsA + buf * ASZ;
    unsigned short* cB = lsB + buf * BSZ;
    if (k0 + 32 < K) {
      issue_tile_async(A, lda, bm, k0 + 32, B, ldb, bn,
                       lsA + (buf ^ 1) * ASZ, lsB + (buf ^ 1) * BSZ);
      // previous tile's 3 ops complete in-order; the 3 just issued stay in flight
      asm volatile("s_wait_asynccnt 3" ::: "memory");
    } else {
      asm volatile("s_wait_asynccnt 0" ::: "memory");
    }
    __syncthreads();

    Frag a[2], b[2];
#pragma unroll
    for (int sm = 0; sm < 2; ++sm) {
      const int r = (mw * 32 + sm * 16 + lr) * 32;
      a[sm].q[0] = *(const uint4*)&cA[r + kb];
      a[sm].q[1] = *(const uint4*)&cA[r + 16 + kb];
    }
#pragma unroll
    for (int sn = 0; sn < 2; ++sn) {
      const int r = (nw * 32 + sn * 16 + lr) * 32;
      b[sn].q[0] = *(const uint4*)&cB[r + kb];
      b[sn].q[1] = *(const uint4*)&cB[r + 16 + kb];
    }
#pragma unroll
    for (int sm = 0; sm < 2; ++sm)
#pragma unroll
      for (int sn = 0; sn < 2; ++sn)
        acc[sm][sn] = __builtin_amdgcn_wmma_f32_16x16x32_bf16(
            false, a[sm].v, false, b[sn].v, (short)0, acc[sm][sn], false, false);
    __syncthreads();   // all waves done reading buf before it is refilled
  }
}

#define GEMM_PROLOGUE()                                            \
  __shared__ unsigned short lsA[2 * ASZ], lsB[2 * BSZ];            \
  const int t    = threadIdx.x;                                    \
  const int lane = t & 31, wv = t >> 5;                            \
  const int mw   = wv & 3,  nw = wv >> 2;                          \
  const int lr   = lane & 15, hi = lane >> 4;                      \
  const int bm = blockIdx.y * BM_T, bn = blockIdx.x * BN_T;        \
  v8f zero = {0.f, 0.f, 0.f, 0.f, 0.f, 0.f, 0.f, 0.f};            \
  v8f acc[2][2];                                                   \
  acc[0][0] = zero; acc[0][1] = zero; acc[1][0] = zero; acc[1][1] = zero; \
  (void)t;

// D element (sm, sn, r): m = bm+mw*32+sm*16+hi*8+r, n = bn+nw*32+sn*16+lr

// ---------------------------------------------------------------------------
// GEMM 1: preact = [x,h] @ Wcat^T (+ zero biases), N padded to 2112
// ---------------------------------------------------------------------------
__global__ void k_gemm_preact(const unsigned short* __restrict__ A,
                              const unsigned short* __restrict__ B,
                              const float* __restrict__ b1,
                              const float* __restrict__ b2,
                              float* __restrict__ C)
{
  GEMM_PROLOGUE();
  gemm_core_128x64(A, KCAT, B, KCAT, KCAT, bm, bn, lsA, lsB, acc);
#pragma unroll
  for (int sm = 0; sm < 2; ++sm)
#pragma unroll
    for (int sn = 0; sn < 2; ++sn)
#pragma unroll
      for (int r = 0; r < 8; ++r) {
        int m = bm + mw * 32 + sm * 16 + hi * 8 + r;
        int n = bn + nw * 32 + sn * 16 + lr;
        float bias = (n < NHT) ? (b1[n] + b2[n]) : 0.f;
        C[(size_t)m * NHT_PAD + n] = acc[sm][sn][r] + bias;
      }
}

// ---------------------------------------------------------------------------
// GEMM 2: ff = inps * (c_t @ vals^T) / max(qn*mn, eps)
// ---------------------------------------------------------------------------
__global__ void k_gemm_sim(const unsigned short* __restrict__ A,
                           const unsigned short* __restrict__ B,
                           const float* __restrict__ qn,
                           const float* __restrict__ mn,
                           const float* __restrict__ inps,
                           float* __restrict__ ff)
{
  GEMM_PROLOGUE();
  gemm_core_128x64(A, H, B, H, H, bm, bn, lsA, lsB, acc);
#pragma unroll
  for (int sm = 0; sm < 2; ++sm)
#pragma unroll
    for (int sn = 0; sn < 2; ++sn)
#pragma unroll
      for (int r = 0; r < 8; ++r) {
        int m = bm + mw * 32 + sm * 16 + hi * 8 + r;
        int n = bn + nw * 32 + sn * 16 + lr;
        float denom = fmaxf(qn[m] * mn[n], EPSV);
        ff[(size_t)m * DICT + n] = inps[m] * acc[sm][sn][r] / denom;
      }
}

// ---------------------------------------------------------------------------
// GEMM 3: m_t = w @ vals ; fused cm_t, h_t, new_vals scatter, bf16 h_t
// ---------------------------------------------------------------------------
__global__ void k_gemm_mt(const unsigned short* __restrict__ A,
                          const unsigned short* __restrict__ B,
                          const float* __restrict__ ct,
                          const float* __restrict__ preact,
                          const int* __restrict__ mp,
                          float* __restrict__ o_mt,
                          float* __restrict__ o_cm,
                          float* __restrict__ o_ht,
                          float* __restrict__ o_nv,
                          unsigned short* __restrict__ htbf)
{
  GEMM_PROLOGUE();
  gemm_core_128x64(A, DICT, B, DICT, DICT, bm, bn, lsA, lsB, acc);
  const int mp0 = mp[0];
#pragma unroll
  for (int sm = 0; sm < 2; ++sm)
#pragma unroll
    for (int sn = 0; sn < 2; ++sn)
#pragma unroll
      for (int r = 0; r < 8; ++r) {
        int m = bm + mw * 32 + sm * 16 + hi * 8 + r;
        int n = bn + nw * 32 + sn * 16 + lr;
        float mt = acc[sm][sn][r];
        float cm = ct[(size_t)m * H + n] + mt;
        float ot = sigf(preact[(size_t)m * NHT_PAD + H + n]);
        float ht = ot * tanhf(cm);
        size_t e = (size_t)m * H + n;
        o_mt[e] = mt;
        o_cm[e] = cm;
        o_ht[e] = ht;
        htbf[e] = f2bf(ht);
        int row = (mp0 + m) % DICT;
        o_nv[(size_t)row * H + n] = cm;
      }
}

// ---------------------------------------------------------------------------
// GEMM 4: a2c = relu(h_t @ W_ih^T + b_ih)
// ---------------------------------------------------------------------------
__global__ void k_gemm_a2c(const unsigned short* __restrict__ A,
                           const unsigned short* __restrict__ B,
                           const float* __restrict__ bih,
                           float* __restrict__ a2c)
{
  GEMM_PROLOGUE();
  gemm_core_128x64(A, H, B, H, H, bm, bn, lsA, lsB, acc);
#pragma unroll
  for (int sm = 0; sm < 2; ++sm)
#pragma unroll
    for (int sn = 0; sn < 2; ++sn)
#pragma unroll
      for (int r = 0; r < 8; ++r) {
        int m = bm + mw * 32 + sm * 16 + hi * 8 + r;
        int n = bn + nw * 32 + sn * 16 + lr;
        a2c[(size_t)m * H + n] = fmaxf(acc[sm][sn][r] + bih[n], 0.f);
      }
}

// ---------------------------------------------------------------------------
// Prep kernels
// ---------------------------------------------------------------------------
__global__ void k_prep_vals(const float* __restrict__ vals,
                            unsigned short* __restrict__ vbf,
                            unsigned short* __restrict__ vbfT,
                            float* __restrict__ mn)
{
  __shared__ float red[256];
  int d = blockIdx.x, t = threadIdx.x;
  float s = 0.f;
  for (int k = t; k < H; k += 256) {
    float v = vals[(size_t)d * H + k];
    s += v * v;
    unsigned short b = f2bf(v);
    vbf[(size_t)d * H + k]   = b;
    vbfT[(size_t)k * DICT + d] = b;   // B-layout for m_t = w @ vals
  }
  red[t] = s; __syncthreads();
  for (int off = 128; off > 0; off >>= 1) {
    if (t < off) red[t] += red[t + off];
    __syncthreads();
  }
  if (t == 0) mn[d] = sqrtf(red[0]);
}

__global__ void k_prep_xh(const float* __restrict__ x,
                          const float* __restrict__ h,
                          unsigned short* __restrict__ xh)
{
  size_t i = (size_t)blockIdx.x * 256 + threadIdx.x;
  int b = (int)(i >> 10), k = (int)(i & 1023);
  float v = (k < DIN) ? x[(size_t)b * DIN + k] : h[(size_t)b * H + (k - DIN)];
  xh[i] = f2bf(v);
}

__global__ void k_prep_wcat(const float* __restrict__ Wi,
                            const float* __restrict__ Wh,
                            unsigned short* __restrict__ Wcat)
{
  size_t i = (size_t)blockIdx.x * 256 + threadIdx.x;
  int n = (int)(i >> 10), k = (int)(i & 1023);
  float v = 0.f;
  if (n < NHT) v = (k < DIN) ? Wi[(size_t)n * DIN + k] : Wh[(size_t)n * H + (k - DIN)];
  Wcat[i] = f2bf(v);
}

__global__ void k_cvt_bf16(const float* __restrict__ src,
                           unsigned short* __restrict__ dst, int n)
{
  int i = blockIdx.x * 256 + threadIdx.x;
  if (i < n) dst[i] = f2bf(src[i]);
}

__global__ void k_copy_vals(const float4* __restrict__ src, float4* __restrict__ dst)
{
  size_t i = (size_t)blockIdx.x * 256 + threadIdx.x;
  dst[i] = src[i];
}

// ---------------------------------------------------------------------------
// Activations: gates, c_t, qn, LCA scalars
// ---------------------------------------------------------------------------
__global__ void k_act(const float* __restrict__ preact,
                      const float* __restrict__ c_prev,
                      float* __restrict__ ct, unsigned short* __restrict__ ctbf,
                      float* __restrict__ qn, float* __restrict__ inps,
                      float* __restrict__ leak, float* __restrict__ comp)
{
  __shared__ float red[256];
  int b = blockIdx.x, t = threadIdx.x;
  const float* pb = preact + (size_t)b * NHT_PAD;
  float s = 0.f;
  for (int j = t; j < H; j += 256) {
    float f  = sigf(pb[j]);
    float it = sigf(pb[2 * H + j]);
    float cn = tanhf(pb[3 * H + 3 + j]);
    float c  = c_prev[(size_t)b * H + j] * f + it * cn;
    ct[(size_t)b * H + j]   = c;
    ctbf[(size_t)b * H + j] = f2bf(c);
    s += c * c;
  }
  red[t] = s; __syncthreads();
  for (int off = 128; off > 0; off >>= 1) {
    if (t < off) red[t] += red[t + off];
    __syncthreads();
  }
  if (t == 0) {
    qn[b]   = sqrtf(red[0]);
    inps[b] = sigf(pb[3 * H]);
    leak[b] = sigf(pb[3 * H + 1]);
    comp[b] = sigf(pb[3 * H + 2]);
  }
}

// ---------------------------------------------------------------------------
// LCA: all 10 cycles fused; V and ff live in registers, row-sum via LDS.
// ---------------------------------------------------------------------------
__global__ void __launch_bounds__(1024)
k_lca(const float* __restrict__ ff,
      const float* __restrict__ leak_a, const float* __restrict__ comp_a,
      unsigned short* __restrict__ wbf)
{
  __shared__ float red[1024];
  int b = blockIdx.x, t = threadIdx.x;
  float lk = leak_a[b], cp = comp_a[b];
  float f[8], V[8];
#pragma unroll
  for (int i = 0; i < 8; ++i) {
    f[i] = ff[(size_t)b * DICT + t + i * 1024];
    V[i] = 0.f;
  }
  for (int c = 0; c < NCYC; ++c) {
    float s = 0.f;
#pragma unroll
    for (int i = 0; i < 8; ++i) s += V[i];
    red[t] = s; __syncthreads();
    for (int off = 512; off > 0; off >>= 1) {
      if (t < off) red[t] += red[t + off];
      __syncthreads();
    }
    float S = red[0];
    __syncthreads();
#pragma unroll
    for (int i = 0; i < 8; ++i) {
      float inhib = cp * (S - V[i]);
      float v = V[i] + DTT * (f[i] - lk * V[i] - inhib);
      V[i] = v > 0.f ? v : 0.f;
    }
  }
#pragma unroll
  for (int i = 0; i < 8; ++i)
    wbf[(size_t)b * DICT + t + i * 1024] = f2bf(V[i]);
}

// ---------------------------------------------------------------------------
// Actor/critic heads: 17 tiny dots + softmax(16); one block per row.
// ---------------------------------------------------------------------------
__global__ void k_head(const float* __restrict__ a2c,
                       const float* __restrict__ Wa, const float* __restrict__ ba,
                       const float* __restrict__ Wc, const float* __restrict__ bc,
                       float* __restrict__ pi, float* __restrict__ value)
{
  __shared__ float sh[512];
  __shared__ float red[256];
  __shared__ float logits[17];
  int b = blockIdx.x, t = threadIdx.x;
  sh[t]       = a2c[(size_t)b * H + t];
  sh[t + 256] = a2c[(size_t)b * H + t + 256];
  __syncthreads();
  for (int j = 0; j < 17; ++j) {
    const float* w = (j < 16) ? (Wa + (size_t)j * H) : Wc;
    float s = sh[t] * w[t] + sh[t + 256] * w[t + 256];
    red[t] = s; __syncthreads();
    for (int off = 128; off > 0; off >>= 1) {
      if (t < off) red[t] += red[t + off];
      __syncthreads();
    }
    if (t == 0) logits[j] = red[0] + ((j < 16) ? ba[j] : bc[0]);
    __syncthreads();
  }
  if (t == 0) {
    float mx = logits[0];
    for (int j = 1; j < 16; ++j) mx = fmaxf(mx, logits[j]);
    float e[16], se = 0.f;
    for (int j = 0; j < 16; ++j) { e[j] = __expf(logits[j] - mx); se += e[j]; }
    for (int j = 0; j < 16; ++j) pi[(size_t)b * 16 + j] = e[j] / se;
    value[b] = logits[16];
  }
}

// ---------------------------------------------------------------------------
extern "C" void kernel_launch(void* const* d_in, const int* in_sizes, int n_in,
                              void* d_out, int out_size, void* d_ws, size_t ws_size,
                              hipStream_t stream)
{
  const float* x    = (const float*)d_in[0];
  const float* h    = (const float*)d_in[1];
  const float* cpv  = (const float*)d_in[2];
  const float* Wi   = (const float*)d_in[3];
  const float* bi   = (const float*)d_in[4];
  const float* Wh   = (const float*)d_in[5];
  const float* bh   = (const float*)d_in[6];
  const float* vals = (const float*)d_in[7];
  const float* Wih  = (const float*)d_in[8];
  const float* bih  = (const float*)d_in[9];
  const float* Wa   = (const float*)d_in[10];
  const float* ba   = (const float*)d_in[11];
  const float* Wc   = (const float*)d_in[12];
  const float* bc   = (const float*)d_in[13];
  const int*   mp   = (const int*)d_in[14];

  float* out   = (float*)d_out;
  float* o_pi  = out;                         // 4096*16
  float* o_val = out + 65536;                 // 4096
  float* o_ht  = out + 69632;                 // 4096*512
  float* o_cm  = o_ht + 2097152;
  float* o_mt  = o_cm + 2097152;
  float* o_nv  = o_mt + 2097152;              // 8192*512

  char* wsp = (char*)d_ws;
  auto alloc = [&](size_t bytes) {
    char* p = wsp;
    wsp += (bytes + 255) & ~(size_t)255;
    return p;
  };
  unsigned short* vbf   = (unsigned short*)alloc((size_t)DICT * H * 2);
  unsigned short* vbfT  = (unsigned short*)alloc((size_t)DICT * H * 2);
  float*          mn    = (float*)alloc((size_t)DICT * 4);
  unsigned short* xh    = (unsigned short*)alloc((size_t)BATCH * KCAT * 2);
  unsigned short* Wcat  = (unsigned short*)alloc((size_t)NHT_PAD * KCAT * 2);
  unsigned short* Wihb  = (unsigned short*)alloc((size_t)H * H * 2);
  float*          preact= (float*)alloc((size_t)BATCH * NHT_PAD * 4);
  float*          ct    = (float*)alloc((size_t)BATCH * H * 4);
  unsigned short* ctbf  = (unsigned short*)alloc((size_t)BATCH * H * 2);
  float*          qn    = (float*)alloc((size_t)BATCH * 4);
  float*          inps  = (float*)alloc((size_t)BATCH * 4);
  float*          leak  = (float*)alloc((size_t)BATCH * 4);
  float*          comp  = (float*)alloc((size_t)BATCH * 4);
  float*          ff    = (float*)alloc((size_t)BATCH * DICT * 4);
  unsigned short* wlca  = (unsigned short*)alloc((size_t)BATCH * DICT * 2);
  unsigned short* htbf  = (unsigned short*)alloc((size_t)BATCH * H * 2);
  float*          a2c   = (float*)alloc((size_t)BATCH * H * 4);

  // independent prep work
  k_prep_vals<<<DICT, 256, 0, stream>>>(vals, vbf, vbfT, mn);
  k_prep_xh<<<(BATCH * KCAT) / 256, 256, 0, stream>>>(x, h, xh);
  k_prep_wcat<<<(NHT_PAD * KCAT) / 256, 256, 0, stream>>>(Wi, Wh, Wcat);
  k_cvt_bf16<<<(H * H) / 256, 256, 0, stream>>>(Wih, Wihb, H * H);
  k_copy_vals<<<(DICT * H / 4) / 256, 256, 0, stream>>>((const float4*)vals,
                                                        (float4*)o_nv);

  // main pipeline
  k_gemm_preact<<<dim3(NHT_PAD / BN_T, BATCH / BM_T), 256, 0, stream>>>(
      xh, Wcat, bi, bh, preact);
  k_act<<<BATCH, 256, 0, stream>>>(preact, cpv, ct, ctbf, qn, inps, leak, comp);
  k_gemm_sim<<<dim3(DICT / BN_T, BATCH / BM_T), 256, 0, stream>>>(
      ctbf, vbf, qn, mn, inps, ff);
  k_lca<<<BATCH, 1024, 0, stream>>>(ff, leak, comp, wlca);
  k_gemm_mt<<<dim3(H / BN_T, BATCH / BM_T), 256, 0, stream>>>(
      wlca, vbfT, ct, preact, mp, o_mt, o_cm, o_ht, o_nv, htbf);
  k_gemm_a2c<<<dim3(H / BN_T, BATCH / BM_T), 256, 0, stream>>>(htbf, Wihb, bih, a2c);
  k_head<<<BATCH, 256, 0, stream>>>(a2c, Wa, ba, Wc, bc, o_pi, o_val);
}
